// ConditionedStableKoopmanTransition_30305289240706
// MI455X (gfx1250) — matrix operand order repaired and checked by reading, not compiled, not run on hardware
//
#include <hip/hip_runtime.h>
#include <hip/hip_bf16.h>
#include <stdint.h>

// ---------------- problem dims ----------------
#define DD   256   // dynamic dim
#define UD   48    // u dim
#define UDP  64    // u dim padded to multiple of 32
#define NO   80    // n_obs
#define TT   128   // rollout steps
#define BB   2048  // batch
#define MT   16    // batch rows per workgroup

typedef __bf16 v16bf __attribute__((ext_vector_type(16)));
typedef __bf16 v8bf  __attribute__((ext_vector_type(8)));
typedef float  v8f   __attribute__((ext_vector_type(8)));

__device__ __forceinline__ unsigned short f2bf(float f) {
    unsigned int u = __float_as_uint(f);
    unsigned int r = u + 0x7FFFu + ((u >> 16) & 1u);   // round-to-nearest-even
    return (unsigned short)(r >> 16);
}

// A-fragment (16x32 bf16, M x K) from LDS, row stride rs elements.
// Lanes 0-15: row m=lane,   K = {k0..k0+7, k0+16..k0+23}
// Lanes16-31: row m=lane-16,K = {k0+8..k0+15, k0+24..k0+31}
__device__ __forceinline__ v16bf loadA(const unsigned short* base, int rs, int k0, int lane) {
    int m  = lane & 15;
    int hi = lane >> 4;
    const unsigned short* p = base + m * rs + k0 + hi * 8;
    v8bf lo = *(const v8bf*)p;          // ds_load_b128
    v8bf h  = *(const v8bf*)(p + 16);   // ds_load_b128
    v16bf r;
#pragma unroll
    for (int i = 0; i < 8; i++) { r[i] = lo[i]; r[i + 8] = h[i]; }
    return r;
}

// B-fragment (32x16 bf16, K x N) where B[j,n] = W[n, j] (W row-major).
// Lanes 0-15: K = {k0..k0+7, k0+16..k0+23}; lanes 16-31: K = {k0+8..k0+15, k0+24..k0+31}
__device__ __forceinline__ v16bf loadB(const unsigned short* rowp, int k0, int laneHi) {
    const unsigned short* p = rowp + k0 + laneHi * 8;
    v8bf lo = *(const v8bf*)p;         // ds_load_b128
    v8bf h  = *(const v8bf*)(p + 16);  // ds_load_b128
    v16bf r;
#pragma unroll
    for (int i = 0; i < 8; i++) { r[i] = lo[i]; r[i + 8] = h[i]; }
    return r;
}

__device__ __forceinline__ v8f wmma_bf16(v16bf a, v16bf b, v8f c) {
    return __builtin_amdgcn_wmma_f32_16x16x32_bf16(false, a, false, b, (short)0, c, false, false);
}

// prefetch one timestep of U into registers (4 values per thread, 16x64 padded tile)
__device__ __forceinline__ void loadU(float* ureg, const float* __restrict__ U,
                                      int t, int brow, int tid) {
#pragma unroll
    for (int j = 0; j < 4; j++) {
        int i = tid + j * 256;
        int r = i >> 6, c = i & 63;
        float v = 0.0f;
        if (t < TT && c < UD) v = U[((size_t)t * BB + brow + r) * UD + c];
        ureg[j] = v;
    }
}

// ---------------- precompute kernels ----------------

__global__ __launch_bounds__(256) void k_init_aug(const float* __restrict__ V,
                                                  float* __restrict__ aug) {
    int idx = blockIdx.x * 256 + threadIdx.x;
    if (idx >= 256 * 512) return;
    int r = idx >> 9, c = idx & 511;
    aug[idx] = (c < 256) ? V[r * 256 + c] : ((c - 256 == r) ? 1.0f : 0.0f);
}

// In-place Gauss-Jordan; Vinv ends up in aug[:, 256:512]. Single block; thread = row.
__global__ __launch_bounds__(256) void k_gauss_jordan(float* __restrict__ aug) {
    __shared__ float prow[512];
    int tid = threadIdx.x;
    for (int p = 0; p < 256; p++) {
        float inv = 1.0f / aug[p * 512 + p];
        __syncthreads();
        for (int c = tid; c < 512; c += 256) {
            float v = aug[p * 512 + c] * inv;
            prow[c] = v;
            aug[p * 512 + c] = v;
        }
        __syncthreads();
        if (tid != p) {
            float f = aug[tid * 512 + p];
            for (int c = 0; c < 512; c++) aug[tid * 512 + c] -= f * prow[c];
        }
        __syncthreads();
    }
}

// Kbf[i][j] = bf16( sum_k V[i][k] * (0.99*sigmoid(eig[k])) * Vinv[k][j] )
__global__ __launch_bounds__(256) void k_build_K(const float* __restrict__ V,
                                                 const float* __restrict__ eig_raw,
                                                 const float* __restrict__ aug,
                                                 unsigned short* __restrict__ Kbf) {
    int idx = blockIdx.x * 256 + threadIdx.x;
    if (idx >= 256 * 256) return;
    int i = idx >> 8, j = idx & 255;
    float s = 0.0f;
    for (int k = 0; k < 256; k++) {
        float e = 0.99f / (1.0f + __expf(-eig_raw[k]));
        s = fmaf(V[i * 256 + k] * e, aug[k * 512 + 256 + j], s);
    }
    Kbf[idx] = f2bf(s);
}

__global__ __launch_bounds__(256) void k_conv_pad(const float* __restrict__ src,
                                                  unsigned short* __restrict__ dst,
                                                  int rows, int cols, int colsp) {
    int idx = blockIdx.x * 256 + threadIdx.x;
    if (idx >= rows * colsp) return;
    int r = idx / colsp, c = idx % colsp;
    dst[idx] = f2bf((c < cols) ? src[r * cols + c] : 0.0f);
}

// ---------------- main rollout kernel ----------------
__global__ __launch_bounds__(256) void k_koopman_roll(
        const float* __restrict__ z_dyn, const float* __restrict__ U,
        const float* __restrict__ dt,
        const unsigned short* __restrict__ Kbf, const unsigned short* __restrict__ Lbf,
        const unsigned short* __restrict__ Cbf, const unsigned short* __restrict__ Dbf,
        float* __restrict__ Z, float* __restrict__ Y) {
    // weights resident in LDS (bf16): 128K+32K+40K+10K; double-buffered state+u: 20K. ~230KB of 320KB.
    __shared__ __attribute__((aligned(16))) unsigned short sK[DD * DD];
    __shared__ __attribute__((aligned(16))) unsigned short sL[DD * UDP];
    __shared__ __attribute__((aligned(16))) unsigned short sC[NO * DD];
    __shared__ __attribute__((aligned(16))) unsigned short sD[NO * UDP];
    __shared__ __attribute__((aligned(16))) unsigned short sz[2 * MT * DD];
    __shared__ __attribute__((aligned(16))) unsigned short su[2 * MT * UDP];

    const int tid    = threadIdx.x;          // 0..255, 8 waves
    const int w      = tid >> 5;
    const int lane   = tid & 31;
    const int laneHi = lane >> 4;
    const int lcol   = lane & 15;
    const int brow   = blockIdx.x * MT;
    const float dts  = dt[0];

    // stage weights global -> LDS (128-bit chunks)
    {
        const uint4* s;
        uint4* d;
        s = (const uint4*)Kbf; d = (uint4*)sK;
        for (int i = tid; i < (DD * DD) / 8; i += 256) d[i] = s[i];
        s = (const uint4*)Lbf; d = (uint4*)sL;
        for (int i = tid; i < (DD * UDP) / 8; i += 256) d[i] = s[i];
        s = (const uint4*)Cbf; d = (uint4*)sC;
        for (int i = tid; i < (NO * DD) / 8; i += 256) d[i] = s[i];
        s = (const uint4*)Dbf; d = (uint4*)sD;
        for (int i = tid; i < (NO * UDP) / 8; i += 256) d[i] = s[i];
    }
    // init bf16 state tile into buffer 0
    for (int i = tid; i < MT * DD; i += 256) {
        int r = i >> 8, c = i & 255;
        sz[i] = f2bf(z_dyn[(size_t)(brow + r) * DD + c]);
    }

    float ureg[4];
    loadU(ureg, U, 0, brow, tid);      // prefetch u_0 (overlaps LDS staging)
    __syncthreads();

    const int n0 = w, n1 = w + 8;      // this wave's two z column-tiles
    const int col0 = (n0 << 4) + lcol;
    const int col1 = (n1 << 4) + lcol;

    for (int t = 0; t < TT; t++) {
        const unsigned short* szc = sz + (t & 1) * (MT * DD);        // current state
        unsigned short*       szn = sz + ((t & 1) ^ 1) * (MT * DD);  // next state
        unsigned short*       suc = su + (t & 1) * (MT * UDP);

        // publish u_t * dt -> bf16 LDS (from registers), then prefetch u_{t+1}
#pragma unroll
        for (int j = 0; j < 4; j++) suc[tid + j * 256] = f2bf(ureg[j] * dts);
        loadU(ureg, U, t + 1, brow, tid);   // global loads overlap the whole compute phase
        __syncthreads();                    // B1: u_t visible; prev y-phase complete

        // z_new = z @ K^T + u' @ L^T   (two 16x16 tiles per wave)
        v8f acc0 = {}, acc1 = {};
        for (int k0 = 0; k0 < DD; k0 += 32) {
            v16bf a  = loadA(szc, DD, k0, lane);
            v16bf b0 = loadB(&sK[(size_t)(n0 * 16 + lcol) * DD], k0, laneHi);
            v16bf b1 = loadB(&sK[(size_t)(n1 * 16 + lcol) * DD], k0, laneHi);
            acc0 = wmma_bf16(a, b0, acc0);
            acc1 = wmma_bf16(a, b1, acc1);
        }
        for (int k0 = 0; k0 < UDP; k0 += 32) {
            v16bf a  = loadA(suc, UDP, k0, lane);
            v16bf b0 = loadB(&sL[(size_t)(n0 * 16 + lcol) * UDP], k0, laneHi);
            v16bf b1 = loadB(&sL[(size_t)(n1 * 16 + lcol) * UDP], k0, laneHi);
            acc0 = wmma_bf16(a, b0, acc0);
            acc1 = wmma_bf16(a, b1, acc1);
        }

        // write Z and the double-buffered next state (no WAR barrier needed)
#pragma unroll
        for (int i = 0; i < 8; i++) {
            int m = (laneHi << 3) + i;     // D layout: lanes0-15 -> M=i, lanes16-31 -> M=8+i
            size_t zoff = ((size_t)t * BB + brow + m) * DD;
            Z[zoff + col0] = acc0[i];
            Z[zoff + col1] = acc1[i];
            szn[m * DD + col0] = f2bf(acc0[i]);
            szn[m * DD + col1] = f2bf(acc1[i]);
        }
        __syncthreads();                    // B2: new state visible

        // y = z_new @ C^T + u' @ D^T  (five 16x16 tiles, waves 0..4)
        if (w < 5) {
            v8f accy = {};
            const unsigned short* crow = &sC[(size_t)(w * 16 + lcol) * DD];
            for (int k0 = 0; k0 < DD; k0 += 32) {
                v16bf a = loadA(szn, DD, k0, lane);
                v16bf b = loadB(crow, k0, laneHi);
                accy = wmma_bf16(a, b, accy);
            }
            const unsigned short* drow = &sD[(size_t)(w * 16 + lcol) * UDP];
            for (int k0 = 0; k0 < UDP; k0 += 32) {
                v16bf a = loadA(suc, UDP, k0, lane);
                v16bf b = loadB(drow, k0, laneHi);
                accy = wmma_bf16(a, b, accy);
            }
            int colY = (w << 4) + lcol;
#pragma unroll
            for (int i = 0; i < 8; i++) {
                int m = (laneHi << 3) + i;
                Y[((size_t)t * BB + brow + m) * NO + colY] = accy[i];
            }
        }
        // no end-of-loop barrier: next iteration writes the *other* su/sz buffers;
        // B1 of the next iteration orders everything else.
    }
}

// ---------------- host launch ----------------
extern "C" void kernel_launch(void* const* d_in, const int* in_sizes, int n_in,
                              void* d_out, int out_size, void* d_ws, size_t ws_size,
                              hipStream_t stream) {
    const float* z_dyn   = (const float*)d_in[0];
    // d_in[1] = z_static (unused by reference)
    const float* dt      = (const float*)d_in[2];
    const float* U       = (const float*)d_in[3];
    const float* eig_raw = (const float*)d_in[4];
    const float* V       = (const float*)d_in[5];
    const float* L       = (const float*)d_in[6];
    const float* C       = (const float*)d_in[7];
    const float* Dm      = (const float*)d_in[8];

    char* ws = (char*)d_ws;
    float*          aug = (float*)(ws);                       // 256*512*4 = 512 KB
    unsigned short* Kbf = (unsigned short*)(ws + 524288);     // 128 KB
    unsigned short* Lbf = (unsigned short*)(ws + 655360);     // 32 KB
    unsigned short* Cbf = (unsigned short*)(ws + 688128);     // 40 KB
    unsigned short* Dbf = (unsigned short*)(ws + 729088);     // 10 KB

    float* Z = (float*)d_out;
    float* Y = Z + (size_t)TT * BB * DD;

    k_init_aug    <<<512, 256, 0, stream>>>(V, aug);
    k_gauss_jordan<<<1,   256, 0, stream>>>(aug);
    k_build_K     <<<256, 256, 0, stream>>>(V, eig_raw, aug, Kbf);
    k_conv_pad    <<<(DD * UDP + 255) / 256, 256, 0, stream>>>(L,  Lbf, DD, UD, UDP);
    k_conv_pad    <<<(NO * DD  + 255) / 256, 256, 0, stream>>>(C,  Cbf, NO, DD, DD);
    k_conv_pad    <<<(NO * UDP + 255) / 256, 256, 0, stream>>>(Dm, Dbf, NO, UD, UDP);

    k_koopman_roll<<<BB / MT, 256, 0, stream>>>(z_dyn, U, dt, Kbf, Lbf, Cbf, Dbf, Z, Y);
}
